// SelfAttentionLayer_35244501630992
// MI455X (gfx1250) — matrix operand order repaired and checked by reading, not compiled, not run on hardware
//
#include <hip/hip_runtime.h>
#include <hip/hip_bf16.h>

// Self-attention (B=2,S=2048,D=1024,H=16,depth=64) for gfx1250 (MI455X).
// fp32 -> bf16, matmuls via v_wmma_f32_16x16x32_bf16 (fp32 acc), flash softmax.
// Attention K/V tiles staged into LDS with global_load_async_to_lds_b128
// (ASYNCcnt, double-buffered); V tile transposed cooperatively in LDS so all
// WMMA fragments are contiguous LDS reads.

#define BB 2
#define SS 2048
#define DD 1024
#define HH 16
#define DH 64
#define NROWS (BB * SS)

typedef __attribute__((ext_vector_type(16))) __bf16 v16bf;
typedef __attribute__((ext_vector_type(8)))  __bf16 v8bf;
typedef __attribute__((ext_vector_type(8)))  float  v8f;

static __device__ __forceinline__ v8bf ld8(const __bf16* p) {
  return *(const v8bf*)p;
}
static __device__ __forceinline__ v16bf cat16(v8bf lo, v8bf hi) {
  return __builtin_shufflevector(lo, hi, 0,1,2,3,4,5,6,7,8,9,10,11,12,13,14,15);
}
static __device__ __forceinline__ v8f wmma_bf16(v16bf a, v16bf b, v8f c) {
  return __builtin_amdgcn_wmma_f32_16x16x32_bf16(false, a, false, b, (short)0, c,
                                                 false, false);
}

// 16B global -> LDS async copy (per-lane); tracked by ASYNCcnt.
static __device__ __forceinline__ void async_copy_b128(unsigned lds_off,
                                                       const void* gaddr) {
  asm volatile("global_load_async_to_lds_b128 %0, %1, off"
               :: "v"(lds_off), "v"(gaddr)
               : "memory");
}
static __device__ __forceinline__ void wait_async0() {
  asm volatile("s_wait_asynccnt 0x0" ::: "memory");
}

// ---------------- conversion kernels ----------------
__global__ void __launch_bounds__(256)
cvt_bf16_kernel(const float* __restrict__ src, __bf16* __restrict__ dst, int n) {
  int i = blockIdx.x * 256 + threadIdx.x;
  if (i < n) dst[i] = (__bf16)src[i];
}

// W (DD x DD row-major, fp32) -> Wt (column-major, bf16): Wt[n*DD + k] = W[k*DD + n]
__global__ void __launch_bounds__(256)
transpose_cvt_kernel(const float* __restrict__ W, __bf16* __restrict__ Wt) {
  int idx = blockIdx.x * 256 + threadIdx.x;   // coalesced read
  int k = idx >> 10;
  int n = idx & (DD - 1);
  Wt[(size_t)n * DD + k] = (__bf16)W[idx];
}

// ---------------- GEMM: X(N x D) @ W(D x D) + b, scatter to (B,H,S,64) bf16 ----
__global__ void __launch_bounds__(256)
gemm_qkv_kernel(const __bf16* __restrict__ X, const __bf16* __restrict__ Wt,
                const float* __restrict__ bias, __bf16* __restrict__ outBHSD) {
  const int wave = threadIdx.x >> 5, lane = threadIdx.x & 31;
  const int lh = lane >> 4, ln = lane & 15;
  const int mTile = blockIdx.y * 8 + wave;
  const int nTile = blockIdx.x;
  const int col = nTile * 16 + ln;
  const __bf16* arow = X  + (size_t)(mTile * 16 + ln) * DD;
  const __bf16* bcol = Wt + (size_t)col * DD;

  v8f acc = {};
  for (int k0 = 0; k0 < DD; k0 += 32) {
    v16bf a = cat16(ld8(arow + k0 + 16 * lh), ld8(arow + k0 + 16 * lh + 8));
    v16bf b = *(const v16bf*)(bcol + k0 + 16 * lh);
    acc = wmma_bf16(a, b, acc);
  }
  const float bc = bias[col];
  const int h = col >> 6, dh = col & (DH - 1);
#pragma unroll
  for (int r = 0; r < 8; ++r) {
    int gm = mTile * 16 + r + 8 * lh;        // global row = b*S + s
    int b_ = gm >> 11, s = gm & (SS - 1);
    outBHSD[((size_t)(b_ * HH + h) * SS + s) * DH + dh] = (__bf16)(acc[r] + bc);
  }
}

// ---------------- GEMM: A(N x D) @ Wo(D x D) + bo -> fp32 out ----------------
__global__ void __launch_bounds__(256)
gemm_out_kernel(const __bf16* __restrict__ X, const __bf16* __restrict__ Wt,
                const float* __restrict__ bias, float* __restrict__ out) {
  const int wave = threadIdx.x >> 5, lane = threadIdx.x & 31;
  const int lh = lane >> 4, ln = lane & 15;
  const int mTile = blockIdx.y * 8 + wave;
  const int nTile = blockIdx.x;
  const int col = nTile * 16 + ln;
  const __bf16* arow = X  + (size_t)(mTile * 16 + ln) * DD;
  const __bf16* bcol = Wt + (size_t)col * DD;

  v8f acc = {};
  for (int k0 = 0; k0 < DD; k0 += 32) {
    v16bf a = cat16(ld8(arow + k0 + 16 * lh), ld8(arow + k0 + 16 * lh + 8));
    v16bf b = *(const v16bf*)(bcol + k0 + 16 * lh);
    acc = wmma_bf16(a, b, acc);
  }
  const float bc = bias[col];
#pragma unroll
  for (int r = 0; r < 8; ++r) {
    int gm = mTile * 16 + r + 8 * lh;
    out[(size_t)gm * DD + col] = acc[r] + bc;
  }
}

// ---------------- flash attention per (b, h, 16-query tile) ----------------
// Block = 8 waves sharing one (b,h). K/V 32x64 tiles async-staged into LDS
// (double-buffered); V transposed in LDS so all fragments are contiguous.
__global__ void __launch_bounds__(256)
attention_kernel(const __bf16* __restrict__ Q, const __bf16* __restrict__ K,
                 const __bf16* __restrict__ V, const unsigned char* __restrict__ mask,
                 __bf16* __restrict__ O) {
  __shared__ __align__(16) __bf16 Ktile[2][32][DH];  // 2 x 4 KB
  __shared__ __align__(16) __bf16 Vtile[2][32][DH];  // 2 x 4 KB
  __shared__ __align__(16) __bf16 VtT[DH][32];       // transposed V, 4 KB
  __shared__ __align__(16) __bf16 Plds[8][16][32];   // per-wave P tile, 8 KB

  const int tid  = threadIdx.x;
  const int wave = tid >> 5, lane = tid & 31;
  const int lh = lane >> 4, ln = lane & 15;
  const int b = blockIdx.z, h = blockIdx.y;
  const int qBase = (blockIdx.x * 8 + wave) * 16;
  const size_t headOff = (size_t)(b * HH + h) * SS * DH;
  const __bf16* Qp = Q + headOff;
  const __bf16* Kp = K + headOff;
  const __bf16* Vp = V + headOff;
  const unsigned char* mk = mask + (size_t)b * SS;

  // cooperative staging map: thread -> (row 0..31, 16B chunk 0..7)
  const int srow = tid >> 3;
  const int schk = tid & 7;

  // Q fragments (16 queries x 64 depth) kept in registers, reused every step.
  v16bf qa0, qa1;
  {
    const __bf16* q = Qp + (size_t)(qBase + ln) * DH;
    qa0 = cat16(ld8(q + 16 * lh),      ld8(q + 16 * lh + 8));
    qa1 = cat16(ld8(q + 32 + 16 * lh), ld8(q + 40 + 16 * lh));
  }

  // prologue: async-stage key block 0 into buffer 0
  {
    unsigned lk = (unsigned)(size_t)&Ktile[0][srow][schk * 8];
    unsigned lv = (unsigned)(size_t)&Vtile[0][srow][schk * 8];
    async_copy_b128(lk, Kp + (size_t)srow * DH + schk * 8);
    async_copy_b128(lv, Vp + (size_t)srow * DH + schk * 8);
  }

  v8f oacc[4] = {{}, {}, {}, {}};
  float rmax[8], rsum[8];
#pragma unroll
  for (int r = 0; r < 8; ++r) { rmax[r] = -3.0e38f; rsum[r] = 0.f; }

  for (int kb = 0; kb < SS; kb += 32) {
    const int cur = (kb >> 5) & 1;
    const int nxt = cur ^ 1;

    wait_async0();        // my async stores to LDS have landed
    __syncthreads();      // everyone's landed; prior reads of 'nxt' & VtT done

    if (kb + 32 < SS) {   // prefetch next key block into the other buffer
      unsigned lk = (unsigned)(size_t)&Ktile[nxt][srow][schk * 8];
      unsigned lv = (unsigned)(size_t)&Vtile[nxt][srow][schk * 8];
      const size_t gofs = (size_t)(kb + 32 + srow) * DH + schk * 8;
      async_copy_b128(lk, Kp + gofs);
      async_copy_b128(lv, Vp + gofs);
    }

    // cooperative transpose: VtT[dh][key] = Vtile[cur][key][dh]
    {
      v8bf vrow = *(const v8bf*)&Vtile[cur][srow][schk * 8];
#pragma unroll
      for (int j = 0; j < 8; ++j) VtT[schk * 8 + j][srow] = vrow[j];
    }
    __syncthreads();      // VtT ready for all waves

    // logits: two 16x16 tiles (keys kb..kb+15, kb+16..kb+31), K-dim = 64
    v8f c0 = {}, c1 = {};
    {
      const __bf16* k0p = &Ktile[cur][ln][0];
      const __bf16* k1p = &Ktile[cur][16 + ln][0];
      c0 = wmma_bf16(qa0, *(const v16bf*)(k0p + 16 * lh), c0);
      c0 = wmma_bf16(qa1, *(const v16bf*)(k0p + 32 + 16 * lh), c0);
      c1 = wmma_bf16(qa0, *(const v16bf*)(k1p + 16 * lh), c1);
      c1 = wmma_bf16(qa1, *(const v16bf*)(k1p + 32 + 16 * lh), c1);
    }
    const float neg0 = mk[kb + ln]      ? 0.f : -1e9f;
    const float neg1 = mk[kb + 16 + ln] ? 0.f : -1e9f;

#pragma unroll
    for (int r = 0; r < 8; ++r) {
      float l0 = c0[r] * 0.125f + neg0;   // 1/sqrt(64)
      float l1 = c1[r] * 0.125f + neg1;
      float mx = fmaxf(l0, l1);
      mx = fmaxf(mx, __shfl_xor(mx, 1, 32));
      mx = fmaxf(mx, __shfl_xor(mx, 2, 32));
      mx = fmaxf(mx, __shfl_xor(mx, 4, 32));
      mx = fmaxf(mx, __shfl_xor(mx, 8, 32));
      float mnew = fmaxf(rmax[r], mx);
      float fsc  = __expf(rmax[r] - mnew);
      rmax[r] = mnew;
      float p0 = __expf(l0 - mnew);
      float p1 = __expf(l1 - mnew);
      float ps = p0 + p1;
      ps += __shfl_xor(ps, 1, 32);
      ps += __shfl_xor(ps, 2, 32);
      ps += __shfl_xor(ps, 4, 32);
      ps += __shfl_xor(ps, 8, 32);
      rsum[r] = rsum[r] * fsc + ps;
      oacc[0][r] *= fsc; oacc[1][r] *= fsc; oacc[2][r] *= fsc; oacc[3][r] *= fsc;
      int m = r + 8 * lh;                  // C row this lane-half owns
      Plds[wave][m][ln]      = (__bf16)p0;
      Plds[wave][m][16 + ln] = (__bf16)p1;
    }
    asm volatile("" ::: "memory");         // same-wave LDS ops are in-order

    // P (16x32) as A-fragment from LDS
    v16bf pa;
    {
      const __bf16* pp = &Plds[wave][ln][0];
      pa = cat16(*(const v8bf*)(pp + 16 * lh), *(const v8bf*)(pp + 16 * lh + 8));
    }
    // O += P @ V(32 x 64): V fragments are contiguous rows of VtT
#pragma unroll
    for (int t = 0; t < 4; ++t) {
      v16bf vb = *(const v16bf*)(&VtT[16 * t + ln][0] + 16 * lh);
      oacc[t] = wmma_bf16(pa, vb, oacc[t]);
    }
  }

  // normalize and write (B,S,D) bf16 attention output
#pragma unroll
  for (int r = 0; r < 8; ++r) {
    float inv = 1.0f / rsum[r];
    int m = r + 8 * lh;
    int s = qBase + m;
    __bf16* op = O + (size_t)(b * SS + s) * DD + h * DH + ln;
    op[0]  = (__bf16)(oacc[0][r] * inv);
    op[16] = (__bf16)(oacc[1][r] * inv);
    op[32] = (__bf16)(oacc[2][r] * inv);
    op[48] = (__bf16)(oacc[3][r] * inv);
  }
}

// ---------------- launch ----------------
extern "C" void kernel_launch(void* const* d_in, const int* in_sizes, int n_in,
                              void* d_out, int out_size, void* d_ws, size_t ws_size,
                              hipStream_t stream) {
  const float* batch = (const float*)d_in[0];
  const unsigned char* mask = (const unsigned char*)d_in[1];  // jax bool = 1 byte
  const float* Wq = (const float*)d_in[2];
  const float* bq = (const float*)d_in[3];
  const float* Wk = (const float*)d_in[4];
  const float* bk = (const float*)d_in[5];
  const float* Wv = (const float*)d_in[6];
  const float* bv = (const float*)d_in[7];
  const float* Wo = (const float*)d_in[8];
  const float* bo = (const float*)d_in[9];
  float* out = (float*)d_out;

  // workspace partition (needs ~48 MB)
  char* ws = (char*)d_ws;
  const size_t MB = (size_t)1 << 20;
  __bf16* Xbf = (__bf16*)(ws + 0);        // 8 MB  batch bf16
  __bf16* Wtq = (__bf16*)(ws + 8  * MB);  // 2 MB  Wq^T bf16
  __bf16* Wtk = (__bf16*)(ws + 10 * MB);
  __bf16* Wtv = (__bf16*)(ws + 12 * MB);
  __bf16* Wto = (__bf16*)(ws + 14 * MB);
  __bf16* Qb  = (__bf16*)(ws + 16 * MB);  // 8 MB  (B,H,S,64)
  __bf16* Kb  = (__bf16*)(ws + 24 * MB);
  __bf16* Vb  = (__bf16*)(ws + 32 * MB);
  __bf16* Ab  = (__bf16*)(ws + 40 * MB);  // 8 MB  attention out (B,S,D)

  cvt_bf16_kernel<<<(NROWS * DD) / 256, 256, 0, stream>>>(batch, Xbf, NROWS * DD);
  transpose_cvt_kernel<<<(DD * DD) / 256, 256, 0, stream>>>(Wq, Wtq);
  transpose_cvt_kernel<<<(DD * DD) / 256, 256, 0, stream>>>(Wk, Wtk);
  transpose_cvt_kernel<<<(DD * DD) / 256, 256, 0, stream>>>(Wv, Wtv);
  transpose_cvt_kernel<<<(DD * DD) / 256, 256, 0, stream>>>(Wo, Wto);

  dim3 ggrid(DD / 16, NROWS / (16 * 8));      // 64 x 32 blocks, 8 waves each
  gemm_qkv_kernel<<<ggrid, 256, 0, stream>>>(Xbf, Wtq, bq, Qb);
  gemm_qkv_kernel<<<ggrid, 256, 0, stream>>>(Xbf, Wtk, bk, Kb);
  gemm_qkv_kernel<<<ggrid, 256, 0, stream>>>(Xbf, Wtv, bv, Vb);

  dim3 agrid(SS / (16 * 8), HH, BB);          // 16 x 16 x 2
  attention_kernel<<<agrid, 256, 0, stream>>>(Qb, Kb, Vb, mask, Ab);

  gemm_out_kernel<<<ggrid, 256, 0, stream>>>(Ab, Wto, bo, out);
}
